// Model_7816840478911
// MI455X (gfx1250) — matrix operand order repaired and checked by reading, not compiled, not run on hardware
//
#include <hip/hip_runtime.h>
#include <hip/hip_bf16.h>
#include <stdint.h>

typedef _Float16 half_t;
typedef __attribute__((ext_vector_type(16))) _Float16 v16h;
typedef __attribute__((ext_vector_type(8)))  _Float16 v8h;
typedef __attribute__((ext_vector_type(8)))  float    v8f;

#define BS      16384
#define NDW     300
#define VOCAB   258
#define EMB     512
#define HID     1024
#define INDIM   812     // EMB + NDW
#define KX      832     // INDIM padded to multiple of 32
#define NV      272     // VOCAB padded to multiple of 16
#define NSTEPS  14

#define SLABK   64      // K columns staged per slab
#define BROWS   64      // 4 gates * 16 weight rows staged per block
#define LDSROW  72      // halves per LDS slab row (64 + 8 pad -> conflict-free)

#if defined(__has_builtin)
#if __has_builtin(__builtin_amdgcn_s_wait_asynccnt)
#define ASYNC_WAIT(n) __builtin_amdgcn_s_wait_asynccnt(n)
#endif
#endif
#ifndef ASYNC_WAIT
#define ASYNC_WAIT(n) asm volatile("s_wait_asynccnt %0" :: "n"(n) : "memory")
#endif

// ---------------------------------------------------------------- helpers ---

__device__ __forceinline__ float sigmoid_(float x) {
    return 1.0f / (1.0f + __expf(-x));
}
__device__ __forceinline__ float tanh_(float x) {
    float e = __expf(2.0f * x);
    return (e - 1.0f) / (e + 1.0f);
}

// A fragment: 16x32 f16 (M x K), rows of `base` (row stride ldk halves).
// lane<16: M=lane, K = kk..kk+15 ; lane>=16: M=lane-16, K = kk+16..kk+31
__device__ __forceinline__ v16h load_fragA(const half_t* __restrict__ base,
                                           int ldk, int kk, int lane) {
    int m  = lane & 15;
    int kh = (lane >> 4) << 4;                 // 0 or 16
    const half_t* p = base + (size_t)m * ldk + kk + kh;
    v8h lo = *(const v8h*)(p);
    v8h hi = *(const v8h*)(p + 8);
    return __builtin_shufflevector(lo, hi, 0,1,2,3,4,5,6,7,8,9,10,11,12,13,14,15);
}

// B fragment: 32x16 f16 (K x N) with B = W^T, W row-major [N][ldk].
// lane<16: N=lane, K = kk..kk+15 ; lane>=16: N=lane-16, K = kk+16..kk+31
__device__ __forceinline__ v16h load_fragB(const half_t* __restrict__ W,
                                           int nbase, int ldk, int kk, int lane) {
    int n  = lane & 15;
    int kh = (lane >> 4) << 4;                 // 0 or 16
    const half_t* p = W + (size_t)(nbase + n) * ldk + kk + kh;
    v8h lo = *(const v8h*)(p);
    v8h hi = *(const v8h*)(p + 8);
    return __builtin_shufflevector(lo, hi, 0,1,2,3,4,5,6,7,8,9,10,11,12,13,14,15);
}

__device__ __forceinline__ v8f wmma_(v16h a, v16h b, v8f c) {
    return __builtin_amdgcn_wmma_f32_16x16x32_f16(
        /*neg_a=*/false, a, /*neg_b=*/false, b,
        /*c_mod=*/(short)0, c, /*reuse_a=*/false, /*reuse_b=*/false);
}

// --------------------------------------------------- kernel A: gates + cell ---
// grid: (BS/128, HID/16), block: 256 (8 waves). Wave w: rows Mbase..Mbase+15,
// 4 WMMA tiles (i,f,g,o) at the same 16 hidden columns j16..j16+15.
// B (weight) tiles are staged into LDS via async global->LDS copies,
// double-buffered in SLABK-column slabs, shared by all 8 waves.
__global__ __launch_bounds__(256) void k_gates_cell(
    const half_t* __restrict__ xt,     // [BS][KX]
    const half_t* __restrict__ h_in,   // [BS][HID]
    const half_t* __restrict__ W_ih,   // [4*HID][KX]  f16, zero-padded K
    const half_t* __restrict__ W_hh,   // [4*HID][HID] f16
    const float*  __restrict__ bias,   // [4*HID] = b_ih + b_hh
    float*        __restrict__ c,      // [BS][HID]  in-place state
    half_t*       __restrict__ h_out)  // [BS][HID]
{
    __shared__ __align__(16) half_t sB[2][BROWS * LDSROW];

    int tid  = threadIdx.x;
    int wave = tid >> 5;
    int lane = tid & 31;
    int Mbase = blockIdx.x * 128 + wave * 16;
    int j16   = blockIdx.y * 16;

    // Issue one slab's async loads: 64 rows x 64 halves = 512 x 16B chunks,
    // 2 chunks per thread -> per-wave ASYNCcnt += 2.
    auto issue_slab = [&](const half_t* __restrict__ W, int ldk, int ks, int buf) {
        #pragma unroll
        for (int i = 0; i < 2; ++i) {
            int q  = tid + i * 256;          // chunk id 0..511
            int r  = q >> 3;                 // staged row 0..63
            int cc = (q & 7) * 8;            // halves within slab row
            int g  = r >> 4;                 // gate
            int n  = r & 15;                 // column within j-tile
            const half_t* gp = W + (size_t)(g * HID + j16 + n) * ldk + ks + cc;
            uint32_t loff = (uint32_t)(uintptr_t)(&sB[buf][r * LDSROW + cc]);
            asm volatile("global_load_async_to_lds_b128 %0, %1, off"
                         :: "v"(loff), "v"((const void*)gp) : "memory");
        }
    };

    // B fragment from the LDS slab: row = g*16 + n, 16 contiguous halves.
    auto loadB_lds = [&](int buf, int g, int kkrel) -> v16h {
        int n  = lane & 15;
        int kh = (lane >> 4) << 4;           // 0 or 16
        const half_t* p = &sB[buf][(g * 16 + n) * LDSROW + kkrel + kh];
        v8h lo = *(const v8h*)(p);
        v8h hi = *(const v8h*)(p + 8);
        return __builtin_shufflevector(lo, hi, 0,1,2,3,4,5,6,7,8,9,10,11,12,13,14,15);
    };

    v8f acc0 = {}, acc1 = {}, acc2 = {}, acc3 = {};

    // ---- phase 1: xt @ W_ih^T  (K = KX = 13 slabs) ----
    {
        const int NS = KX / SLABK;           // 13
        const half_t* Abase = xt + (size_t)Mbase * KX;
        issue_slab(W_ih, KX, 0, 0);
        for (int s = 0; s < NS; ++s) {
            int ks = s * SLABK;
            if (s + 1 < NS) {
                issue_slab(W_ih, KX, ks + SLABK, (s + 1) & 1);
                ASYNC_WAIT(2);               // slab s complete (in-order)
            } else {
                ASYNC_WAIT(0);
            }
            __syncthreads();                 // slab s visible to all waves
            int buf = s & 1;
            #pragma unroll
            for (int kr = 0; kr < SLABK; kr += 32) {
                // materialize all fragments first so ds_loads clause together
                // and the scheduler can use counted (non-zero) dscnt waits
                v16h a  = load_fragA(Abase, KX, ks + kr, lane);
                v16h b0 = loadB_lds(buf, 0, kr);
                v16h b1 = loadB_lds(buf, 1, kr);
                v16h b2 = loadB_lds(buf, 2, kr);
                v16h b3 = loadB_lds(buf, 3, kr);
                acc0 = wmma_(a, b0, acc0);
                acc1 = wmma_(a, b1, acc1);
                acc2 = wmma_(a, b2, acc2);
                acc3 = wmma_(a, b3, acc3);
            }
            __syncthreads();                 // allow slab s+2 to overwrite buf
        }
    }
    // ---- phase 2: h @ W_hh^T  (K = HID = 16 slabs) ----
    {
        const int NS = HID / SLABK;          // 16
        const half_t* Abase = h_in + (size_t)Mbase * HID;
        issue_slab(W_hh, HID, 0, 0);
        for (int s = 0; s < NS; ++s) {
            int ks = s * SLABK;
            if (s + 1 < NS) {
                issue_slab(W_hh, HID, ks + SLABK, (s + 1) & 1);
                ASYNC_WAIT(2);
            } else {
                ASYNC_WAIT(0);
            }
            __syncthreads();
            int buf = s & 1;
            #pragma unroll
            for (int kr = 0; kr < SLABK; kr += 32) {
                v16h a  = load_fragA(Abase, HID, ks + kr, lane);
                v16h b0 = loadB_lds(buf, 0, kr);
                v16h b1 = loadB_lds(buf, 1, kr);
                v16h b2 = loadB_lds(buf, 2, kr);
                v16h b3 = loadB_lds(buf, 3, kr);
                acc0 = wmma_(a, b0, acc0);
                acc1 = wmma_(a, b1, acc1);
                acc2 = wmma_(a, b2, acc2);
                acc3 = wmma_(a, b3, acc3);
            }
            __syncthreads();
        }
    }

    int n  = lane & 15;
    int hh = lane >> 4;
    int j  = j16 + n;
    float bI = bias[0 * HID + j];
    float bF = bias[1 * HID + j];
    float bG = bias[2 * HID + j];
    float bO = bias[3 * HID + j];

    #pragma unroll
    for (int r = 0; r < 8; ++r) {
        int brow = Mbase + r + (hh << 3);
        size_t idx = (size_t)brow * HID + j;
        float iv = sigmoid_(acc0[r] + bI);
        float fv = sigmoid_(acc1[r] + bF);
        float gv = tanh_  (acc2[r] + bG);
        float ov = sigmoid_(acc3[r] + bO);
        float cn = fv * c[idx] + iv * gv;
        c[idx] = cn;
        h_out[idx] = (half_t)(ov * tanh_(cn));
    }
}

// ----------------------- kernel B: logits + softmax-top1 + embedding gather ---
// grid: BS/16, block 256 (8 waves). All waves share the same 16 batch rows;
// 17 N-tiles of 16 vocab columns distributed over waves.
__global__ __launch_bounds__(256) void k_logits_argmax(
    const half_t* __restrict__ h,      // [BS][HID] (new h)
    const half_t* __restrict__ W_lin,  // [NV][HID] f16, rows >=VOCAB are zero
    const float*  __restrict__ b_lin,  // [VOCAB]
    const half_t* __restrict__ emb_h,  // [VOCAB][EMB]
    half_t*       __restrict__ xt,     // [BS][KX] (overwrite first EMB cols)
    float*        __restrict__ resp,   // [NSTEPS*BS]
    int*          __restrict__ resi,   // [NSTEPS*BS]
    int step)
{
    __shared__ float lg[16][NV];
    __shared__ float rmax[256];
    __shared__ int   rid [256];
    __shared__ float rsum[256];
    __shared__ float rowmax[16];
    __shared__ int   rowidx[16];

    int wave = threadIdx.x >> 5;
    int lane = threadIdx.x & 31;
    int Mbase = blockIdx.x * 16;

    const half_t* Abase = h + (size_t)Mbase * HID;
    int nt0 = wave, nt1 = wave + 8;
    bool three = (wave == 0);                  // wave-uniform -> EXEC stays full
    v8f a0 = {}, a1 = {}, a2 = {};

    #pragma unroll 2
    for (int kk = 0; kk < HID; kk += 32) {
        v16h a  = load_fragA(Abase, HID, kk, lane);
        v16h b0 = load_fragB(W_lin, nt0 * 16, HID, kk, lane);
        v16h b1 = load_fragB(W_lin, nt1 * 16, HID, kk, lane);
        a0 = wmma_(a, b0, a0);
        a1 = wmma_(a, b1, a1);
        if (three) {
            v16h b2 = load_fragB(W_lin, 256, HID, kk, lane);
            a2 = wmma_(a, b2, a2);
        }
    }

    int n  = lane & 15;
    int hh = lane >> 4;
    #pragma unroll
    for (int r = 0; r < 8; ++r) {
        int row = r + (hh << 3);
        lg[row][nt0 * 16 + n] = a0[r];
        lg[row][nt1 * 16 + n] = a1[r];
        if (three) lg[row][256 + n] = a2[r];
    }
    __syncthreads();

    int row = threadIdx.x >> 4;
    int t16 = threadIdx.x & 15;

    // max + argmax over 258 cols
    float best = -3.0e38f;
    int   bidx = 0;
    for (int col = t16; col < VOCAB; col += 16) {
        float v = lg[row][col] + b_lin[col];
        if (v > best) { best = v; bidx = col; }
    }
    rmax[threadIdx.x] = best;
    rid [threadIdx.x] = bidx;
    __syncthreads();
    if (t16 == 0) {
        float m = rmax[threadIdx.x]; int mi = rid[threadIdx.x];
        for (int t = 1; t < 16; ++t) {
            float v = rmax[threadIdx.x + t]; int vi = rid[threadIdx.x + t];
            if (v > m || (v == m && vi < mi)) { m = v; mi = vi; }
        }
        rowmax[row] = m;
        rowidx[row] = mi;
    }
    __syncthreads();

    // sum of exp(logit - max); top-1 prob = 1 / sum
    float mx = rowmax[row];
    float s = 0.0f;
    for (int col = t16; col < VOCAB; col += 16)
        s += __expf(lg[row][col] + b_lin[col] - mx);
    rsum[threadIdx.x] = s;
    __syncthreads();
    if (t16 == 0) {
        float tot = 0.0f;
        for (int t = 0; t < 16; ++t) tot += rsum[threadIdx.x + t];
        int brow = Mbase + row;
        resp[(size_t)step * BS + brow] = 1.0f / tot;
        resi[(size_t)step * BS + brow] = rowidx[row];
    }
    __syncthreads();

    // gather emb[argmax] into xt[:, 0:EMB] for next step (32 halves per thread)
    int idx  = rowidx[row];
    int brow = Mbase + row;
    const v8h* src = (const v8h*)(emb_h + (size_t)idx  * EMB + t16 * 32);
    v8h*       dst = (v8h*)      (xt    + (size_t)brow * KX  + t16 * 32);
    #pragma unroll
    for (int q = 0; q < 4; ++q) dst[q] = src[q];
}

// ------------------------------------------------------------ init kernels ---

__global__ void k_cvt_pad(const float* __restrict__ in, half_t* __restrict__ out,
                          int Nsrc, int Ksrc, int Kp, long total) {
    long i = (long)blockIdx.x * blockDim.x + threadIdx.x;
    if (i >= total) return;
    int nr = (int)(i / Kp);
    int k  = (int)(i % Kp);
    float v = (nr < Nsrc && k < Ksrc) ? in[(size_t)nr * Ksrc + k] : 0.0f;
    out[i] = (half_t)v;
}

__global__ void k_bias(const float* __restrict__ a, const float* __restrict__ b,
                       float* __restrict__ o, int n) {
    int i = blockIdx.x * blockDim.x + threadIdx.x;
    if (i < n) o[i] = a[i] + b[i];
}

__global__ void k_init_state(half_t* __restrict__ h0, half_t* __restrict__ h1,
                             float* __restrict__ c, long total) {
    long i = (long)blockIdx.x * blockDim.x + threadIdx.x;
    if (i >= total) return;
    h0[i] = (half_t)0.0f;
    h1[i] = (half_t)0.0f;
    c[i]  = 0.0f;
}

__global__ void k_init_xt(const float* __restrict__ emb,
                          const float* __restrict__ wv,
                          half_t* __restrict__ xt, long total) {
    long i = (long)blockIdx.x * blockDim.x + threadIdx.x;
    if (i >= total) return;
    int b   = (int)(i / KX);
    int col = (int)(i % KX);
    float v;
    if (col < EMB)        v = emb[col];                       // emb row 0 (<start>)
    else if (col < INDIM) v = wv[(size_t)b * NDW + (col - EMB)];
    else                  v = 0.0f;
    xt[i] = (half_t)v;
}

// ---------------------------------------------------------------- launcher ---

extern "C" void kernel_launch(void* const* d_in, const int* in_sizes, int n_in,
                              void* d_out, int out_size, void* d_ws, size_t ws_size,
                              hipStream_t stream) {
    const float* word_vec = (const float*)d_in[0];
    const float* emb      = (const float*)d_in[1];
    const float* W_ih     = (const float*)d_in[2];
    const float* W_hh     = (const float*)d_in[3];
    const float* b_ih     = (const float*)d_in[4];
    const float* b_hh     = (const float*)d_in[5];
    const float* W_lin    = (const float*)d_in[6];
    const float* b_lin    = (const float*)d_in[7];

    char* p = (char*)d_ws;
    half_t* W_ih_h  = (half_t*)p; p += (size_t)4 * HID * KX  * sizeof(half_t);
    half_t* W_hh_h  = (half_t*)p; p += (size_t)4 * HID * HID * sizeof(half_t);
    half_t* W_lin_h = (half_t*)p; p += (size_t)NV * HID      * sizeof(half_t);
    half_t* emb_h   = (half_t*)p; p += (size_t)VOCAB * EMB   * sizeof(half_t);
    float*  bias    = (float*) p; p += (size_t)4 * HID       * sizeof(float);
    half_t* xt      = (half_t*)p; p += (size_t)BS * KX       * sizeof(half_t);
    half_t* hbuf0   = (half_t*)p; p += (size_t)BS * HID      * sizeof(half_t);
    half_t* hbuf1   = (half_t*)p; p += (size_t)BS * HID      * sizeof(half_t);
    float*  c       = (float*) p; p += (size_t)BS * HID      * sizeof(float);

    float* resp = (float*)d_out;
    int*   resi = (int*)((float*)d_out + (size_t)NSTEPS * BS);

    const int T = 256;
    long tWih = (long)4 * HID * KX;
    k_cvt_pad<<<(unsigned)((tWih + T - 1) / T), T, 0, stream>>>(W_ih, W_ih_h, 4 * HID, INDIM, KX, tWih);
    long tWhh = (long)4 * HID * HID;
    k_cvt_pad<<<(unsigned)((tWhh + T - 1) / T), T, 0, stream>>>(W_hh, W_hh_h, 4 * HID, HID, HID, tWhh);
    long tWl = (long)NV * HID;
    k_cvt_pad<<<(unsigned)((tWl + T - 1) / T), T, 0, stream>>>(W_lin, W_lin_h, VOCAB, HID, HID, tWl);
    long tE = (long)VOCAB * EMB;
    k_cvt_pad<<<(unsigned)((tE + T - 1) / T), T, 0, stream>>>(emb, emb_h, VOCAB, EMB, EMB, tE);
    k_bias<<<(4 * HID + T - 1) / T, T, 0, stream>>>(b_ih, b_hh, bias, 4 * HID);
    long tS = (long)BS * HID;
    k_init_state<<<(unsigned)((tS + T - 1) / T), T, 0, stream>>>(hbuf0, hbuf1, c, tS);
    long tX = (long)BS * KX;
    k_init_xt<<<(unsigned)((tX + T - 1) / T), T, 0, stream>>>(emb, word_vec, xt, tX);

    half_t* hb[2] = { hbuf0, hbuf1 };
    for (int t = 0; t < NSTEPS; ++t) {
        half_t* hin  = hb[t & 1];
        half_t* hout = hb[(t + 1) & 1];
        dim3 gA(BS / 128, HID / 16);
        k_gates_cell<<<gA, 256, 0, stream>>>(xt, hin, W_ih_h, W_hh_h, bias, c, hout);
        k_logits_argmax<<<BS / 16, 256, 0, stream>>>(hout, W_lin_h, b_lin, emb_h,
                                                     xt, resp, resi, t);
    }
}
